// SSCDelaysSNN_15831249453659
// MI455X (gfx1250) — compile-verified, matrix operand order checked by reading, not run on hardware
//
#include <hip/hip_runtime.h>
#include <hip/hip_bf16.h>

// ---------------------------------------------------------------------------
// SSC delays SNN, fused CDNA5 WMMA implementation.
// B=128 T=250 C=700(pad 704) H1=1024 H2=768 O=35(pad 48)
// ---------------------------------------------------------------------------

typedef _Float16 v8h  __attribute__((ext_vector_type(8)));
typedef _Float16 v16h __attribute__((ext_vector_type(16)));
typedef float    v8f  __attribute__((ext_vector_type(8)));

#define Bsz   128
#define Tsz   250
#define Csz   700
#define CPAD  704
#define H1sz  1024
#define H2sz  768
#define Osz   35
#define OPAD  48

// Workspace layout (bytes), all 256-aligned.
#define OFF_WD    ((size_t)0)                              // 1024x704  f16
#define OFF_WR1   ((size_t)1441792)                        // 1024x1024 f16
#define OFF_WF2   ((size_t)3538944)                        // 768x1024  f16
#define OFF_WR2   ((size_t)5111808)                        // 768x768   f16
#define OFF_WO    ((size_t)6291456)                        // 48x768    f16
#define OFF_V1    ((size_t)6365184)                        // 128x1024  f32
#define OFF_A1    ((size_t)6889472)                        // 128x1024  f32
#define OFF_V2    ((size_t)7413760)                        // 128x768   f32
#define OFF_A2    ((size_t)7806976)                        // 128x768   f32
#define OFF_SPK1  ((size_t)8200192)                        // 2x128x1024 f16 (double buffer)
#define OFF_SPK2H ((size_t)8724480)                        // 251x128x768 f16 (history, slot0=zero)

// Load one 16-half A/B fragment for v_wmma_f32_16x16x32_f16.
// Per ISA layout: lane (p,g): halves 0..7 at K = kbase+8g .. +7,
// halves 8..15 at K = kbase+16+8g .. +7.  base must point at row + kbase + 8g.
__device__ __forceinline__ v16h load_frag(const _Float16* __restrict__ base) {
  v8h lo = *(const v8h*)(base);
  v8h hi = *(const v8h*)(base + 16);
  return __builtin_shufflevector(lo, hi, 0,1,2,3,4,5,6,7,8,9,10,11,12,13,14,15);
}

__device__ __forceinline__ v8f wmma_f16(v16h a, v16h b, v8f c) {
  return __builtin_amdgcn_wmma_f32_16x16x32_f16(false, a, false, b, (short)0, c,
                                                false, false);
}

// ---------------------------------------------------------------------------
// Kernel 0: convert weights fp32 -> f16 (with padding), zero all state.
// Idempotent: safe to re-run on every graph replay.
// ---------------------------------------------------------------------------
__global__ void __launch_bounds__(256) snn_init_kernel(
    const float* __restrict__ wd, const float* __restrict__ wr1,
    const float* __restrict__ wf2, const float* __restrict__ wr2,
    const float* __restrict__ wo, char* __restrict__ ws) {
  size_t tid = (size_t)blockIdx.x * blockDim.x + threadIdx.x;
  size_t stride = (size_t)gridDim.x * blockDim.x;

  _Float16* wd16 = (_Float16*)(ws + OFF_WD);
  _Float16* wr1h = (_Float16*)(ws + OFF_WR1);
  _Float16* wf2h = (_Float16*)(ws + OFF_WF2);
  _Float16* wr2h = (_Float16*)(ws + OFF_WR2);
  _Float16* woh  = (_Float16*)(ws + OFF_WO);
  float* v1 = (float*)(ws + OFF_V1);
  float* a1 = (float*)(ws + OFF_A1);
  float* v2 = (float*)(ws + OFF_V2);
  float* a2 = (float*)(ws + OFF_A2);
  _Float16* spk1  = (_Float16*)(ws + OFF_SPK1);
  _Float16* spk2h = (_Float16*)(ws + OFF_SPK2H);

  for (size_t i = tid; i < (size_t)H1sz * CPAD; i += stride) {
    int n = (int)(i / CPAD), k = (int)(i % CPAD);
    wd16[i] = (k < Csz) ? (_Float16)wd[(size_t)n * Csz + k] : (_Float16)0.0f;
  }
  for (size_t i = tid; i < (size_t)H1sz * H1sz; i += stride) wr1h[i] = (_Float16)wr1[i];
  for (size_t i = tid; i < (size_t)H2sz * H1sz; i += stride) wf2h[i] = (_Float16)wf2[i];
  for (size_t i = tid; i < (size_t)H2sz * H2sz; i += stride) wr2h[i] = (_Float16)wr2[i];
  for (size_t i = tid; i < (size_t)OPAD * H2sz; i += stride) {
    int n = (int)(i / H2sz), k = (int)(i % H2sz);
    woh[i] = (n < Osz) ? (_Float16)wo[(size_t)n * H2sz + k] : (_Float16)0.0f;
  }
  for (size_t i = tid; i < (size_t)Bsz * H1sz; i += stride) { v1[i] = 0.0f; a1[i] = 0.0f; }
  for (size_t i = tid; i < (size_t)Bsz * H2sz; i += stride) { v2[i] = 0.0f; a2[i] = 0.0f; }
  for (size_t i = tid; i < (size_t)2 * Bsz * H1sz; i += stride) spk1[i] = (_Float16)0.0f;
  for (size_t i = tid; i < (size_t)Bsz * H2sz; i += stride) spk2h[i] = (_Float16)0.0f; // slot 0
}

// ---------------------------------------------------------------------------
// Kernel 1 (per step): I1 = xt_delayed @ Wd^T + spk1_prev @ Wr1^T, adLIF, spikes.
// Grid (H1/64, B/32), block 256 (8 waves, 2x4 tiles of 16x16).
// ---------------------------------------------------------------------------
__global__ void __launch_bounds__(256) snn_layer1_kernel(
    const float* __restrict__ x, const int* __restrict__ delays,
    const float* __restrict__ alpha, const float* __restrict__ rho,
    const float* __restrict__ beta_a, char* __restrict__ ws, int t) {
  __shared__ __align__(16) _Float16 sx[32 * CPAD];

  const _Float16* wd16 = (const _Float16*)(ws + OFF_WD);
  const _Float16* wr1h = (const _Float16*)(ws + OFF_WR1);
  float* v1 = (float*)(ws + OFF_V1);
  float* a1 = (float*)(ws + OFF_A1);
  _Float16* spk1 = (_Float16*)(ws + OFF_SPK1);
  const _Float16* spk_prev = spk1 + (size_t)((t + 1) & 1) * Bsz * H1sz;
  _Float16*       spk_new  = spk1 + (size_t)(t & 1) * Bsz * H1sz;

  const int wgB0 = blockIdx.y * 32;

  // Gather delayed inputs for this 32-row batch tile into LDS (f16).
  for (int i = threadIdx.x; i < 32 * CPAD; i += 256) {
    int bl = i / CPAD, c = i % CPAD;
    float val = 0.0f;
    if (c < Csz) {
      int tt = t - delays[c];
      if (tt >= 0) val = x[((size_t)(wgB0 + bl) * Tsz + tt) * Csz + c];
    }
    sx[i] = (_Float16)val;
  }
  __syncthreads();

  const int wave = threadIdx.x >> 5;
  const int lane = threadIdx.x & 31;
  const int p = lane & 15, g = lane >> 4;
  const int b0 = wgB0 + (wave & 1) * 16;
  const int n0 = blockIdx.x * 64 + (wave >> 1) * 16;

  v8f acc = {};

  // Phase A: delayed-input GEMM, A fragments from LDS.
  {
    const _Float16* arow = sx + (size_t)((wave & 1) * 16 + p) * CPAD + 8 * g;
    const _Float16* brow = wd16 + (size_t)(n0 + p) * CPAD + 8 * g;
    for (int kb = 0; kb < CPAD; kb += 32)
      acc = wmma_f16(load_frag(arow + kb), load_frag(brow + kb), acc);
  }
  // Phase B: recurrent GEMM, A fragments from f16 spike buffer in global.
  {
    const _Float16* arow = spk_prev + (size_t)(b0 + p) * H1sz + 8 * g;
    const _Float16* brow = wr1h + (size_t)(n0 + p) * H1sz + 8 * g;
    for (int kb = 0; kb < H1sz; kb += 32)
      acc = wmma_f16(load_frag(arow + kb), load_frag(brow + kb), acc);
  }

  // Fused adLIF epilogue.
  const int n = n0 + p;
  const float al = alpha[n], rh = rho[n], ba = beta_a[n];
  #pragma unroll
  for (int r = 0; r < 8; ++r) {
    const int b = b0 + r + 8 * g;
    const size_t idx = (size_t)b * H1sz + n;
    float a = rh * a1[idx] + ba * (float)spk_prev[idx];
    float v = al * v1[idx] + (1.0f - al) * acc[r] - a;
    float s = (v >= 1.0f) ? 1.0f : 0.0f;
    v1[idx] = v - s;
    a1[idx] = a;
    spk_new[idx] = (_Float16)s;
  }
}

// ---------------------------------------------------------------------------
// Kernel 2 (per step): I2 = spk1 @ Wfc2^T + spk2_prev @ Wr2^T, adLIF, spikes
// written into the history buffer (slot t+1 read-free -> no race).
// Grid (H2/64, B/32), block 256.
// ---------------------------------------------------------------------------
__global__ void __launch_bounds__(256) snn_layer2_kernel(
    const float* __restrict__ alpha, const float* __restrict__ rho,
    const float* __restrict__ beta_a, char* __restrict__ ws, int t) {
  const _Float16* wf2h = (const _Float16*)(ws + OFF_WF2);
  const _Float16* wr2h = (const _Float16*)(ws + OFF_WR2);
  float* v2 = (float*)(ws + OFF_V2);
  float* a2 = (float*)(ws + OFF_A2);
  const _Float16* spk1 =
      (const _Float16*)(ws + OFF_SPK1) + (size_t)(t & 1) * Bsz * H1sz;
  _Float16* hist = (_Float16*)(ws + OFF_SPK2H);
  const _Float16* spk_prev = hist + (size_t)t * Bsz * H2sz;
  _Float16*       spk_new  = hist + (size_t)(t + 1) * Bsz * H2sz;

  const int wave = threadIdx.x >> 5;
  const int lane = threadIdx.x & 31;
  const int p = lane & 15, g = lane >> 4;
  const int b0 = blockIdx.y * 32 + (wave & 1) * 16;
  const int n0 = blockIdx.x * 64 + (wave >> 1) * 16;

  v8f acc = {};
  {
    const _Float16* arow = spk1 + (size_t)(b0 + p) * H1sz + 8 * g;
    const _Float16* brow = wf2h + (size_t)(n0 + p) * H1sz + 8 * g;
    for (int kb = 0; kb < H1sz; kb += 32)
      acc = wmma_f16(load_frag(arow + kb), load_frag(brow + kb), acc);
  }
  {
    const _Float16* arow = spk_prev + (size_t)(b0 + p) * H2sz + 8 * g;
    const _Float16* brow = wr2h + (size_t)(n0 + p) * H2sz + 8 * g;
    for (int kb = 0; kb < H2sz; kb += 32)
      acc = wmma_f16(load_frag(arow + kb), load_frag(brow + kb), acc);
  }

  const int n = n0 + p;
  const float al = alpha[n], rh = rho[n], ba = beta_a[n];
  #pragma unroll
  for (int r = 0; r < 8; ++r) {
    const int b = b0 + r + 8 * g;
    const size_t idx = (size_t)b * H2sz + n;
    float a = rh * a2[idx] + ba * (float)spk_prev[idx];
    float v = al * v2[idx] + (1.0f - al) * acc[r] - a;
    float s = (v >= 1.0f) ? 1.0f : 0.0f;
    v2[idx] = v - s;
    a2[idx] = a;
    spk_new[idx] = (_Float16)s;
  }
}

// ---------------------------------------------------------------------------
// Kernel 3: output layer. Each wave owns a 16x16 (batch x out) tile, scans all
// T steps: I_out = spk2[t] @ Wout^T (24 WMMAs), leaky v_out kept in registers.
// Grid (OPAD/16, B/16), block 32.
// ---------------------------------------------------------------------------
__global__ void __launch_bounds__(32) snn_out_kernel(
    const float* __restrict__ beta_out, char* __restrict__ ws,
    float* __restrict__ out) {
  const _Float16* woh  = (const _Float16*)(ws + OFF_WO);
  const _Float16* hist = (const _Float16*)(ws + OFF_SPK2H);

  const int lane = threadIdx.x & 31;
  const int p = lane & 15, g = lane >> 4;
  const int n0 = blockIdx.x * 16;
  const int b0 = blockIdx.y * 16;
  const float beta = beta_out[0];

  float vout[8], osum[8];
  #pragma unroll
  for (int r = 0; r < 8; ++r) { vout[r] = 0.0f; osum[r] = 0.0f; }

  const _Float16* brow = woh + (size_t)(n0 + p) * H2sz + 8 * g;

  for (int t = 0; t < Tsz; ++t) {
    const _Float16* arow =
        hist + (size_t)(t + 1) * Bsz * H2sz + (size_t)(b0 + p) * H2sz + 8 * g;
    v8f acc = {};
    for (int kb = 0; kb < H2sz; kb += 32)
      acc = wmma_f16(load_frag(arow + kb), load_frag(brow + kb), acc);
    #pragma unroll
    for (int r = 0; r < 8; ++r) {
      vout[r] = beta * vout[r] + (1.0f - beta) * acc[r];
      osum[r] += vout[r];
    }
  }

  const int n = n0 + p;
  if (n < Osz) {
    #pragma unroll
    for (int r = 0; r < 8; ++r) {
      const int b = b0 + r + 8 * g;
      out[(size_t)b * Osz + n] = osum[r] * (1.0f / (float)Tsz);
    }
  }
}

// ---------------------------------------------------------------------------
extern "C" void kernel_launch(void* const* d_in, const int* in_sizes, int n_in,
                              void* d_out, int out_size, void* d_ws, size_t ws_size,
                              hipStream_t stream) {
  (void)in_sizes; (void)n_in; (void)out_size; (void)ws_size;
  const float* x      = (const float*)d_in[0];
  const int*   delays = (const int*)d_in[1];
  const float* wd     = (const float*)d_in[2];
  const float* wr1    = (const float*)d_in[3];
  const float* wf2    = (const float*)d_in[4];
  const float* wr2    = (const float*)d_in[5];
  const float* wo     = (const float*)d_in[6];
  const float* alpha1 = (const float*)d_in[7];
  const float* rho1   = (const float*)d_in[8];
  const float* ba1    = (const float*)d_in[9];
  const float* alpha2 = (const float*)d_in[10];
  const float* rho2   = (const float*)d_in[11];
  const float* ba2    = (const float*)d_in[12];
  const float* betao  = (const float*)d_in[13];
  char*  ws  = (char*)d_ws;
  float* out = (float*)d_out;

  snn_init_kernel<<<dim3(512), dim3(256), 0, stream>>>(wd, wr1, wf2, wr2, wo, ws);
  for (int t = 0; t < Tsz; ++t) {
    snn_layer1_kernel<<<dim3(H1sz / 64, Bsz / 32), dim3(256), 0, stream>>>(
        x, delays, alpha1, rho1, ba1, ws, t);
    snn_layer2_kernel<<<dim3(H2sz / 64, Bsz / 32), dim3(256), 0, stream>>>(
        alpha2, rho2, ba2, ws, t);
  }
  snn_out_kernel<<<dim3(OPAD / 16, Bsz / 16), dim3(32), 0, stream>>>(betao, ws, out);
}
